// HPB_11811160064053
// MI455X (gfx1250) — matrix operand order repaired and checked by reading, not compiled, not run on hardware
//
#include <hip/hip_runtime.h>
#include <hip/hip_bf16.h>

// ---------------- types / WMMA helpers ----------------
typedef _Float16 v16h __attribute__((ext_vector_type(16)));
typedef _Float16 v8h  __attribute__((ext_vector_type(8)));
typedef float    v8f  __attribute__((ext_vector_type(8)));
typedef float    v4f  __attribute__((ext_vector_type(4)));
typedef int      v4i  __attribute__((ext_vector_type(4)));

#define DEV __device__ __forceinline__

// ---- optional CDNA5 async global->LDS path (ASYNCcnt), guarded ----
#if defined(__has_builtin)
#  if __has_builtin(__builtin_amdgcn_global_load_async_to_lds_b128) && \
      __has_builtin(__builtin_amdgcn_s_wait_asynccnt)
#    define USE_ASYNC_LDS 1
#  endif
#endif
#ifndef USE_ASYNC_LDS
#  define USE_ASYNC_LDS 0
#endif

#if USE_ASYNC_LDS
typedef __attribute__((address_space(1))) v4i gv4i;   // global int4*
typedef __attribute__((address_space(3))) v4i lv4i;   // LDS int4*
DEV lv4i* to_lds(void* p){ return (lv4i*)(unsigned int)(uintptr_t)p; }
DEV gv4i* to_glb(const void* p){ return (gv4i*)(uintptr_t)p; }
#endif

// A-operand fragment (16x32 f16): two 8-half chunks per lane, contiguous at every call site (k = 16h + s).
DEV v16h frag_ab(const _Float16* p0, const _Float16* p1){
  v8h lo = *(const v8h*)p0, hi = *(const v8h*)p1;
  v16h r;
#pragma unroll
  for (int i=0;i<8;++i){ r[i]=lo[i]; r[i+8]=hi[i]; }
  return r;
}
// B-operand fragment (32x16 f16): 16 contiguous halves per lane (32B aligned).
DEV v16h frag_b(const _Float16* p){ return *(const v16h*)p; }

DEV v8f wmma16(v16h a, v16h b, v8f c){
  return __builtin_amdgcn_wmma_f32_16x16x32_f16(false, a, false, b, (short)0, c, false, false);
}

DEV float gelu_exact(float x){ return 0.5f*x*(1.0f+erff(x*0.7071067811865475f)); }

// ---------------- problem constants ----------------
constexpr int NPOS = 65536;          // B*H*W
constexpr float EPSn = 1e-5f;

// ---------------- workspace arena (bytes) ----------------
constexpr size_t MBs = 1024ull*1024ull;
constexpr size_t OFF_WQKV = 0;                      // 768*256 f16
constexpr size_t OFF_WOUT = 0x60000;                // 256*256 f16
constexpr size_t OFF_WCOMB= 0x80000;                // 256*512 f16
constexpr size_t OFF_WFF1 = 0xC0000;                // 1024*256 f16
constexpr size_t OFF_WFF2 = 0x140000;               // 256*1024 f16
constexpr size_t OFF_XN   = 4*MBs;                  // xn16 [N][256]   (32MB) ; reused as or16
constexpr size_t OFF_OR   = OFF_XN;
constexpr size_t OFF_QKV  = 36*MBs;                 // qkv16 [768][N]  (96MB) ; reused:
constexpr size_t OFF_CAT  = 36*MBs;                 //   cat16 [N][512] (64MB)
constexpr size_t OFF_COMB = 100*MBs;                //   comb16 [N][256] (32MB)
constexpr size_t OFF_S    = 132*MBs;                // small buffers
constexpr size_t OFF_QP   = OFF_S + 0x0;            // 64*32 f32
constexpr size_t OFF_KS1  = OFF_S + 0x10000;        // 64*32*32 f32
constexpr size_t OFF_IH   = OFF_S + 0x60000;        // 64*8 int
constexpr size_t OFF_IW   = OFF_S + 0x61000;        // 64*8 int
constexpr size_t OFF_KS2  = OFF_S + 0x80000;        // 64*256*32 f32
constexpr size_t OFF_K2   = OFF_S + 0x300000;       // 64*64*32 f16
constexpr size_t OFF_V2T  = OFF_S + 0x340000;       // 64*32*64 f16
constexpr size_t OFF_MEAN = OFF_S + 0x400000;       // up to 8192 f32
constexpr size_t OFF_VAR  = OFF_S + 0x410000;
constexpr size_t OFF_D    = 140*MBs;                // h1pre16 (128MB) -> dwo16 -> ypre32(64MB)
constexpr size_t OFF_E    = 268*MBs;                // h1_16 (128MB)
constexpr size_t OFF_F    = 396*MBs;                // h2_16 (128MB)

// ---------------- weight fp32 -> fp16 ----------------
__global__ void wcvt_k(const float* __restrict__ s, _Float16* __restrict__ d, int n){
  int t = blockIdx.x*256 + threadIdx.x;
  if (t < n) d[t] = (_Float16)s[t];
}

// ---------------- channel layernorm -> xn16 [N][256] ----------------
__global__ __launch_bounds__(256) void cln_k(const float* __restrict__ x,
                                             const float* __restrict__ g,
                                             const float* __restrict__ bt,
                                             _Float16* __restrict__ xn){
  __shared__ float rs[256], rq[256];
  int n = blockIdx.x, c = threadIdx.x;
  int b = n>>13, s = n&8191;
  float v = x[((size_t)(b*256+c))*8192 + s];
  rs[c]=v; rq[c]=v*v; __syncthreads();
  for (int o=128;o>0;o>>=1){ if (c<o){ rs[c]+=rs[c+o]; rq[c]+=rq[c+o]; } __syncthreads(); }
  float m = rs[0]*(1.f/256.f);
  float var = rq[0]*(1.f/256.f) - m*m;
  float y = (v-m)*rsqrtf(var+EPSn)*g[c] + bt[c];
  xn[(size_t)n*256 + c] = (_Float16)y;
}

// ---------------- generic WMMA GEMM: D = A[M][K] x B[N][K]^T layout ----------------
// Double-buffered LDS pipeline; async global->LDS when available.
// EPI: 0 plain f16 out; 1 +bias f16; 2 +bias +residual(x NCHW) f16; 3 +bias f32 out
// CMAJOR: store D channel-major [M][NPOS] (qkv), else position-major [n*ldd + cofs + m]
template<int EPI, bool CMAJOR>
__global__ __launch_bounds__(128) void gemm_wmma_k(
    const _Float16* __restrict__ A, const _Float16* __restrict__ B,
    const float* __restrict__ bias, const float* __restrict__ resid,
    _Float16* __restrict__ Dh, float* __restrict__ Df,
    int M, int K, int ldd, int cofs)
{
  __shared__ _Float16 sA[2][64*32];
  __shared__ _Float16 sB[2][64*32];
  const int tid = threadIdx.x, wave = tid>>5, lane = tid&31;
  const int m0 = blockIdx.y*64, n0 = blockIdx.x*64;
  const int row = tid>>1, hf = (tid&1)*16;
  const int ml = lane&15, kg = lane>>4, ko = kg*16;
  v8f acc[4] = {};

#if USE_ASYNC_LDS
  // Each wave DMAs 1KB of A and 1KB of B per tile: 4 lane-wide b128 async ops.
  auto issue = [&](int kk2, int bf){
    const int c0 = wave*64 + lane;               // chunk ids c0, c0+32 in [0,256)
#pragma unroll
    for (int u=0; u<2; ++u){
      const int c = c0 + u*32;
      const int r2 = c>>2, part = (c&3)*8;       // row 0..63, 8-half part
      __builtin_amdgcn_global_load_async_to_lds_b128(
          to_glb(&A[(size_t)(m0+r2)*K + kk2 + part]),
          to_lds(&sA[bf][r2*32 + part]), 0, 0);
      __builtin_amdgcn_global_load_async_to_lds_b128(
          to_glb(&B[(size_t)(n0+r2)*K + kk2 + part]),
          to_lds(&sB[bf][r2*32 + part]), 0, 0);
    }
  };
#else
  v8h ra0, ra1, rb0, rb1;
  auto load_regs = [&](int kk2){
    ra0 = *(const v8h*)&A[(size_t)(m0+row)*K + kk2 + hf];
    ra1 = *(const v8h*)&A[(size_t)(m0+row)*K + kk2 + hf + 8];
    rb0 = *(const v8h*)&B[(size_t)(n0+row)*K + kk2 + hf];
    rb1 = *(const v8h*)&B[(size_t)(n0+row)*K + kk2 + hf + 8];
  };
  auto store_regs = [&](int bf){
    *(v8h*)&sA[bf][row*32+hf]   = ra0;
    *(v8h*)&sA[bf][row*32+hf+8] = ra1;
    *(v8h*)&sB[bf][row*32+hf]   = rb0;
    *(v8h*)&sB[bf][row*32+hf+8] = rb1;
  };
#endif

  int buf = 0;
#if USE_ASYNC_LDS
  issue(0, 0);
#else
  load_regs(0);
#endif
  for (int kk=0; kk<K; kk+=32){
    const bool more = (kk+32) < K;
    if (kk+64 < K) __builtin_prefetch(&B[(size_t)(n0+row)*K + kk + 64], 0, 0);
#if USE_ASYNC_LDS
    if (more) issue(kk+32, buf^1);               // overlap next-tile DMA with math
    if (more) __builtin_amdgcn_s_wait_asynccnt(4);
    else      __builtin_amdgcn_s_wait_asynccnt(0);
#else
    store_regs(buf);
    if (more) load_regs(kk+32);                  // next-tile loads fly during WMMAs
#endif
    __syncthreads();
    const _Float16* pa = &sA[buf][(wave*16+ml)*32 + ko];
    v16h af = frag_ab(pa, pa+8);
#pragma unroll
    for (int nt=0; nt<4; ++nt){
      v16h bfr = frag_b(&sB[buf][(nt*16+ml)*32 + ko]);
      acc[nt] = wmma16(af, bfr, acc[nt]);
    }
    __syncthreads();
    buf ^= 1;
  }

  const int mb = kg*8;
  const int cbase = m0 + wave*16 + mb;
#pragma unroll
  for (int nt=0; nt<4; ++nt){
    const int n = n0 + nt*16 + ml;
    float vals[8];
#pragma unroll
    for (int r=0;r<8;++r) vals[r] = acc[nt][r];
    if (EPI>=1){
#pragma unroll
      for (int r=0;r<8;++r) vals[r] += bias[cbase+r];
    }
    if (EPI==2){
      int b = n>>13, s = n&8191;
#pragma unroll
      for (int r=0;r<8;++r) vals[r] += resid[((size_t)(b*256+cbase+r))*8192 + s];
    }
    if (CMAJOR){
#pragma unroll
      for (int r=0;r<8;++r) Dh[(size_t)(cbase+r)*NPOS + n] = (_Float16)vals[r];
    } else if (EPI==3){
      v4f o0,o1;
#pragma unroll
      for (int r=0;r<4;++r){ o0[r]=vals[r]; o1[r]=vals[r+4]; }
      *(v4f*)&Df[(size_t)n*ldd + cbase]     = o0;
      *(v4f*)&Df[(size_t)n*ldd + cbase + 4] = o1;
    } else {
      v8h o;
#pragma unroll
      for (int r=0;r<8;++r) o[r]=(_Float16)vals[r];
      *(v8h*)&Dh[(size_t)n*ldd + cofs + cbase] = o;
    }
  }
}

// ---------------- l2 normalize q,k rows (in-place over w=32) ----------------
__global__ __launch_bounds__(256) void l2norm_k(_Float16* __restrict__ qkv){
  int t = blockIdx.x*256 + threadIdx.x;        // 512 * 8 * 256 = 1,048,576
  int m = t >> 11, rem = t & 2047, b = rem >> 8, hh = rem & 255;
  _Float16* p = qkv + (size_t)m*NPOS + (size_t)b*8192 + hh*32;
  float ss = 0.f;
  for (int w=0;w<32;++w){ float v=(float)p[w]; ss += v*v; }
  float rn = 1.f / fmaxf(sqrtf(ss), 1e-12f);
  for (int w=0;w<32;++w) p[w] = (_Float16)((float)p[w]*rn);
}

// ---------------- q_probe[bh][w] = sum_{dc,hh} q ----------------
__global__ __launch_bounds__(256) void qprobe_k(const _Float16* __restrict__ qkv, float* __restrict__ qp){
  int bh = blockIdx.x, b = bh>>3, hd = bh&7;
  int w = threadIdx.x & 31, ch = threadIdx.x >> 5;   // 8 hh-chunks
  float s = 0.f;
  for (int dc=0;dc<32;++dc)
    for (int hh=ch*32; hh<ch*32+32; ++hh)
      s += (float)qkv[(size_t)(hd*32+dc)*NPOS + (size_t)b*8192 + hh*32 + w];
  __shared__ float red[256];
  red[threadIdx.x]=s; __syncthreads();
  for (int o=128;o>=32;o>>=1){ if ((int)threadIdx.x<o) red[threadIdx.x]+=red[threadIdx.x+o]; __syncthreads(); }
  if (threadIdx.x<32) qp[bh*32+w] = red[threadIdx.x];
}

// ---------------- ksum1[bh][dc][w] = sum_hh k ----------------
__global__ void ksum1_k(const _Float16* __restrict__ qkv, float* __restrict__ ks){
  int t = blockIdx.x*256 + threadIdx.x;              // 65536
  int w = t&31, dc = (t>>5)&31, bh = t>>10, b = bh>>3, hd = bh&7;
  float s=0.f;
  for (int hh=0;hh<256;++hh)
    s += (float)qkv[(size_t)(256 + hd*32+dc)*NPOS + (size_t)b*8192 + hh*32 + w];
  ks[t] = s;
}

// ---------------- selection 1: top-KH over dc ----------------
__global__ void sel1_k(const float* __restrict__ qp, const float* __restrict__ ks1, int* __restrict__ ih){
  int bh = threadIdx.x; if (bh >= 64) return;
  float sc[32]; bool used[32];
  for (int dc=0;dc<32;++dc){
    float s=0.f;
    for (int w=0;w<32;++w) s += qp[bh*32+w]*ks1[(bh*32+dc)*32 + w];
    sc[dc]=s; used[dc]=false;
  }
  for (int j=0;j<8;++j){
    int bi=0; float bv=-3.4e38f;
    for (int dc=0;dc<32;++dc) if (!used[dc] && sc[dc]>bv){ bv=sc[dc]; bi=dc; }
    used[bi]=true; ih[bh*8+j]=bi;
  }
}

// ---------------- ksum2[bh][hh][w] = sum over selected dc ----------------
__global__ void ksum2_k(const _Float16* __restrict__ qkv, const int* __restrict__ ih, float* __restrict__ ks){
  int t = blockIdx.x*256 + threadIdx.x;              // 524288
  int w = t&31, hh = (t>>5)&255, bh = t>>13, b = bh>>3, hd = bh&7;
  float s=0.f;
  for (int di=0;di<8;++di){
    int dc = ih[bh*8+di];
    s += (float)qkv[(size_t)(256 + hd*32+dc)*NPOS + (size_t)b*8192 + hh*32 + w];
  }
  ks[t] = s;
}

// ---------------- selection 2: top-KW over hh ----------------
__global__ void sel2_k(const float* __restrict__ qp, const float* __restrict__ ks2, int* __restrict__ iw){
  int bh = threadIdx.x; if (bh >= 64) return;
  float sc[256]; bool used[256];
  for (int hh=0;hh<256;++hh){
    float s=0.f;
    for (int w=0;w<32;++w) s += qp[bh*32+w]*ks2[(bh*256+hh)*32 + w];
    sc[hh]=s; used[hh]=false;
  }
  for (int j=0;j<8;++j){
    int bi=0; float bv=-3.4e38f;
    for (int hh=0;hh<256;++hh) if (!used[hh] && sc[hh]>bv){ bv=sc[hh]; bi=hh; }
    used[bi]=true; iw[bh*8+j]=bi;
  }
}

// ---------------- gather k2[bh][64][32], v2t[bh][32][64] ----------------
__global__ void gather_k(const _Float16* __restrict__ qkv, const int* __restrict__ ih,
                         const int* __restrict__ iw, _Float16* __restrict__ k2,
                         _Float16* __restrict__ v2t){
  int t = blockIdx.x*256 + threadIdx.x;              // 131072
  int w = t&31, j = (t>>5)&63, bh = t>>11, b = bh>>3, hd = bh&7;
  int di = j>>3, hi = j&7;
  int dc = ih[bh*8+di], hh = iw[bh*8+hi];
  _Float16 kv = qkv[(size_t)(256 + hd*32+dc)*NPOS + (size_t)b*8192 + hh*32 + w];
  _Float16 vv = qkv[(size_t)(512 + hd*32+dc)*NPOS + (size_t)b*8192 + hh*32 + w];
  k2 [(size_t)bh*2048 + j*32 + w] = kv;
  v2t[(size_t)bh*2048 + w*64 + j] = vv;
}

// ---------------- fused attention (one wave per 16-row block) ----------------
__global__ __launch_bounds__(32) void attn_k(const _Float16* __restrict__ qkv,
                                             const _Float16* __restrict__ k2b,
                                             const _Float16* __restrict__ v2tb,
                                             _Float16* __restrict__ or16){
  __shared__ float    sS[16*64];
  __shared__ _Float16 sP[16*64];
  const int bh = blockIdx.y, i0 = blockIdx.x*16;
  const int b = bh>>3, hd = bh&7;
  const int lane = threadIdx.x;
  const int ml = lane&15, kg = lane>>4, ko = kg*16, mb = kg*8;
  const int i = i0 + ml, dc = i>>8, hh = i&255;
  // S = q2(16x32) @ k2^T : A-frag from channel-major qkv (w contiguous)
  const _Float16* qrow = qkv + (size_t)(hd*32+dc)*NPOS + (size_t)b*8192 + hh*32;
  v16h af = frag_ab(qrow + ko, qrow + ko + 8);
  const _Float16* k2 = k2b + (size_t)bh*2048;
  v8f s4[4] = {};
#pragma unroll
  for (int jt=0;jt<4;++jt){
    v16h bf = frag_b(&k2[(jt*16+ml)*32 + ko]);
    s4[jt] = wmma16(af, bf, s4[jt]);
  }
#pragma unroll
  for (int jt=0;jt<4;++jt)
#pragma unroll
    for (int r=0;r<8;++r)
      sS[(mb+r)*64 + jt*16 + ml] = s4[jt][r];
  __syncthreads();
  if (lane < 16){
    float mx = -3.4e38f;
    for (int j=0;j<64;++j) mx = fmaxf(mx, sS[lane*64+j]);
    float sum = 0.f;
    for (int j=0;j<64;++j){ float e = __expf(sS[lane*64+j]-mx); sS[lane*64+j]=e; sum+=e; }
    float inv = 1.f/sum;
    for (int j=0;j<64;++j) sP[lane*64+j] = (_Float16)(sS[lane*64+j]*inv);
  }
  __syncthreads();
  // O = P(16x64) @ v2(64x32): two K-steps, two N(=w) tiles
  const _Float16* vt = v2tb + (size_t)bh*2048;
  v8f o2[2] = {};
#pragma unroll
  for (int st=0;st<2;++st){
    const _Float16* pp = &sP[ml*64 + st*32 + ko];
    v16h pf = frag_ab(pp, pp+8);
#pragma unroll
    for (int wt=0;wt<2;++wt){
      v16h vf = frag_b(&vt[(wt*16+ml)*64 + st*32 + ko]);
      o2[wt] = wmma16(pf, vf, o2[wt]);
    }
  }
  // fused rearrange store: or16[(b*8192 + i)*256 + hd*32 + d]
#pragma unroll
  for (int wt=0;wt<2;++wt)
#pragma unroll
    for (int r=0;r<8;++r)
      or16[((size_t)b*8192 + i0 + mb + r)*256 + hd*32 + wt*16 + ml] = (_Float16)o2[wt][r];
}

// ---------------- depthwise 3x3 on x -> cat16 cols 256..511 ----------------
__global__ void dwconv_x_k(const float* __restrict__ x, const float* __restrict__ w,
                           const float* __restrict__ bias, _Float16* __restrict__ cat){
  size_t t = (size_t)blockIdx.x*256 + threadIdx.x;   // 256 * 65536
  int c = (int)(t >> 16), n = (int)(t & 65535);
  int b = n>>13, hw = n&8191, h = hw>>5, wq = hw&31;
  float acc = bias[c];
#pragma unroll
  for (int dy=0;dy<3;++dy)
#pragma unroll
    for (int dx=0;dx<3;++dx){
      int hh=h+dy-1, ww=wq+dx-1;
      if (hh>=0 && hh<256 && ww>=0 && ww<32)
        acc += x[(((size_t)(b*256+c))*256 + hh)*32 + ww]*w[c*9+dy*3+dx];
    }
  cat[(size_t)n*512 + 256 + c] = (_Float16)acc;
}

// ---------------- depthwise 3x3 on h1_16 [n][1024] -> dwo16 ----------------
__global__ void ffdw_k(const _Float16* __restrict__ h1, const float* __restrict__ w,
                       const float* __restrict__ bias, _Float16* __restrict__ dwo){
  size_t t = (size_t)blockIdx.x*256 + threadIdx.x;   // 1024 * 65536
  int c = (int)(t & 1023); int n = (int)(t >> 10);
  int b = n>>13, hw = n&8191, h = hw>>5, wq = hw&31;
  float acc = bias[c];
#pragma unroll
  for (int dy=0;dy<3;++dy)
#pragma unroll
    for (int dx=0;dx<3;++dx){
      int hh=h+dy-1, ww=wq+dx-1;
      if (hh>=0 && hh<256 && ww>=0 && ww<32)
        acc += (float)h1[((size_t)b*8192 + hh*32 + ww)*1024 + c]*w[c*9+dy*3+dx];
    }
  dwo[t] = (_Float16)acc;
}

// ---------------- instance norm statistics per (b,c) over 8192 spatial ----------------
template<typename T>
__global__ __launch_bounds__(256) void inorm_stats_k(const T* __restrict__ inp,
                                                     float* __restrict__ mean,
                                                     float* __restrict__ var, int C){
  int g = blockIdx.x, b = g / C, c = g - b*C;
  float s=0.f, q=0.f;
  for (int sp=threadIdx.x; sp<8192; sp+=256){
    float v = (float)inp[((size_t)(b*8192+sp))*C + c];
    s += v; q += v*v;
  }
  __shared__ float rs[256], rq[256];
  rs[threadIdx.x]=s; rq[threadIdx.x]=q; __syncthreads();
  for (int o=128;o>0;o>>=1){
    if ((int)threadIdx.x<o){ rs[threadIdx.x]+=rs[threadIdx.x+o]; rq[threadIdx.x]+=rq[threadIdx.x+o]; }
    __syncthreads();
  }
  if (threadIdx.x==0){
    float m = rs[0]*(1.f/8192.f);
    mean[g]=m; var[g]=rq[0]*(1.f/8192.f)-m*m;
  }
}

// ---------------- h1 = gelu(inorm(ff1)) ----------------
__global__ void h1_apply_k(const _Float16* __restrict__ pre, const float* __restrict__ mean,
                           const float* __restrict__ var, _Float16* __restrict__ h1){
  size_t t = (size_t)blockIdx.x*256 + threadIdx.x;
  int c = (int)(t & 1023); int b = (int)((t >> 10) >> 13);
  int g = b*1024 + c;
  float v = ((float)pre[t] - mean[g])*rsqrtf(var[g]+EPSn);
  h1[t] = (_Float16)gelu_exact(v);
}

// ---------------- h2 = h1 + gelu(inorm(ffdw(h1))) ----------------
__global__ void h2_apply_k(const _Float16* __restrict__ h1, const _Float16* __restrict__ dwo,
                           const float* __restrict__ mean, const float* __restrict__ var,
                           _Float16* __restrict__ h2){
  size_t t = (size_t)blockIdx.x*256 + threadIdx.x;
  int c = (int)(t & 1023); int b = (int)((t >> 10) >> 13);
  int g = b*1024 + c;
  float v = ((float)dwo[t] - mean[g])*rsqrtf(var[g]+EPSn);
  h2[t] = (_Float16)((float)h1[t] + gelu_exact(v));
}

// ---------------- final: inorm(ff2) -> d_out (NCHW f32) ----------------
__global__ void final_apply_k(const float* __restrict__ y, const float* __restrict__ mean,
                              const float* __restrict__ var, float* __restrict__ out){
  size_t t = (size_t)blockIdx.x*256 + threadIdx.x;   // 256 * 65536
  int c = (int)(t & 255); int n = (int)(t >> 8);
  int b = n>>13, s = n&8191;
  int g = b*256 + c;
  out[((size_t)(b*256+c))*8192 + s] = (y[t] - mean[g])*rsqrtf(var[g]+EPSn);
}

// ---------------- launch ----------------
extern "C" void kernel_launch(void* const* d_in, const int* in_sizes, int n_in,
                              void* d_out, int out_size, void* d_ws, size_t ws_size,
                              hipStream_t stream)
{
  (void)in_sizes; (void)n_in; (void)out_size; (void)ws_size;
  const float* x      = (const float*)d_in[0];
  const float* cln_g  = (const float*)d_in[1];
  const float* cln_b  = (const float*)d_in[2];
  const float* w_qkv  = (const float*)d_in[3];
  const float* w_out  = (const float*)d_in[4];
  const float* b_out  = (const float*)d_in[5];
  const float* w_dw   = (const float*)d_in[6];
  const float* b_dw   = (const float*)d_in[7];
  const float* w_comb = (const float*)d_in[8];
  const float* b_comb = (const float*)d_in[9];
  const float* w_ff1  = (const float*)d_in[10];
  const float* b_ff1  = (const float*)d_in[11];
  const float* w_ffdw = (const float*)d_in[12];
  const float* b_ffdw = (const float*)d_in[13];
  const float* w_ff2  = (const float*)d_in[14];
  const float* b_ff2  = (const float*)d_in[15];
  float* out = (float*)d_out;
  char* ws = (char*)d_ws;

  _Float16* wqkv16 = (_Float16*)(ws + OFF_WQKV);
  _Float16* wout16 = (_Float16*)(ws + OFF_WOUT);
  _Float16* wcomb16= (_Float16*)(ws + OFF_WCOMB);
  _Float16* wff116 = (_Float16*)(ws + OFF_WFF1);
  _Float16* wff216 = (_Float16*)(ws + OFF_WFF2);
  _Float16* xn16   = (_Float16*)(ws + OFF_XN);
  _Float16* or16   = (_Float16*)(ws + OFF_OR);
  _Float16* qkv16  = (_Float16*)(ws + OFF_QKV);
  _Float16* cat16  = (_Float16*)(ws + OFF_CAT);
  _Float16* comb16 = (_Float16*)(ws + OFF_COMB);
  float*    qp     = (float*)(ws + OFF_QP);
  float*    ks1    = (float*)(ws + OFF_KS1);
  int*      ih     = (int*)(ws + OFF_IH);
  int*      iw     = (int*)(ws + OFF_IW);
  float*    ks2    = (float*)(ws + OFF_KS2);
  _Float16* k2     = (_Float16*)(ws + OFF_K2);
  _Float16* v2t    = (_Float16*)(ws + OFF_V2T);
  float*    meanb  = (float*)(ws + OFF_MEAN);
  float*    varb   = (float*)(ws + OFF_VAR);
  _Float16* h1pre  = (_Float16*)(ws + OFF_D);
  _Float16* dwo16  = (_Float16*)(ws + OFF_D);
  float*    ypre   = (float*)(ws + OFF_D);
  _Float16* h1_16  = (_Float16*)(ws + OFF_E);
  _Float16* h2_16  = (_Float16*)(ws + OFF_F);

  // weights -> fp16
  wcvt_k<<<(196608+255)/256,256,0,stream>>>(w_qkv, wqkv16, 196608);
  wcvt_k<<<(65536 +255)/256,256,0,stream>>>(w_out, wout16, 65536);
  wcvt_k<<<(131072+255)/256,256,0,stream>>>(w_comb, wcomb16, 131072);
  wcvt_k<<<(262144+255)/256,256,0,stream>>>(w_ff1, wff116, 262144);
  wcvt_k<<<(262144+255)/256,256,0,stream>>>(w_ff2, wff216, 262144);

  // channel LN
  cln_k<<<NPOS,256,0,stream>>>(x, cln_g, cln_b, xn16);

  // qkv = Wqkv * xn  (768 x N, channel-major out)
  gemm_wmma_k<0,true><<<dim3(NPOS/64, 768/64),128,0,stream>>>(
      wqkv16, xn16, nullptr, nullptr, qkv16, nullptr, 768, 256, 0, 0);

  // l2 normalize q,k rows (0..511) over w
  l2norm_k<<<4096,256,0,stream>>>(qkv16);

  // probe + selections + gather
  qprobe_k<<<64,256,0,stream>>>(qkv16, qp);
  ksum1_k <<<256,256,0,stream>>>(qkv16, ks1);
  sel1_k  <<<1,64,0,stream>>>(qp, ks1, ih);
  ksum2_k <<<2048,256,0,stream>>>(qkv16, ih, ks2);
  sel2_k  <<<1,64,0,stream>>>(qp, ks2, iw);
  gather_k<<<512,256,0,stream>>>(qkv16, ih, iw, k2, v2t);

  // fused attention + rearrange -> or16 [N][256]
  attn_k<<<dim3(512,64),32,0,stream>>>(qkv16, k2, v2t, or16);

  // attn branch projection -> cat16 cols 0..255
  gemm_wmma_k<1,false><<<dim3(NPOS/64, 256/64),128,0,stream>>>(
      wout16, or16, b_out, nullptr, cat16, nullptr, 256, 256, 512, 0);

  // conv branch -> cat16 cols 256..511
  dwconv_x_k<<<NPOS,256,0,stream>>>(x, w_dw, b_dw, cat16);

  // combine + residual x -> comb16 [N][256]
  gemm_wmma_k<2,false><<<dim3(NPOS/64, 256/64),128,0,stream>>>(
      wcomb16, cat16, b_comb, x, comb16, nullptr, 256, 512, 256, 0);

  // ff1 -> h1pre [N][1024]
  gemm_wmma_k<1,false><<<dim3(NPOS/64, 1024/64),128,0,stream>>>(
      wff116, comb16, b_ff1, nullptr, h1pre, nullptr, 1024, 256, 1024, 0);

  // inorm + gelu -> h1_16
  inorm_stats_k<_Float16><<<8192,256,0,stream>>>(h1pre, meanb, varb, 1024);
  h1_apply_k<<<262144,256,0,stream>>>(h1pre, meanb, varb, h1_16);

  // ffdw(h1) -> dwo16 (reuses h1pre region)
  ffdw_k<<<262144,256,0,stream>>>(h1_16, w_ffdw, b_ffdw, dwo16);

  // h2 = h1 + gelu(inorm(dwo))
  inorm_stats_k<_Float16><<<8192,256,0,stream>>>(dwo16, meanb, varb, 1024);
  h2_apply_k<<<262144,256,0,stream>>>(h1_16, dwo16, meanb, varb, h2_16);

  // ff2 -> ypre [N][256] f32 (reuses OFF_D region)
  gemm_wmma_k<3,false><<<dim3(NPOS/64, 256/64),128,0,stream>>>(
      wff216, h2_16, b_ff2, nullptr, nullptr, ypre, 256, 1024, 256, 0);

  // final instance norm -> d_out (NCHW fp32)
  inorm_stats_k<float><<<2048,256,0,stream>>>(ypre, meanb, varb, 256);
  final_apply_k<<<NPOS,256,0,stream>>>(ypre, meanb, varb, out);
}
